// Attention_38628935860433
// MI455X (gfx1250) — matrix operand order repaired and checked by reading, not compiled, not run on hardware
//
#include <hip/hip_runtime.h>
#include <hip/hip_bf16.h>

// ---------------------------------------------------------------------------
// Types for CDNA5 WMMA (wave32): v16bf A/B operands, v8f accumulator
// ---------------------------------------------------------------------------
typedef __attribute__((ext_vector_type(16))) __bf16 v16bf;
typedef __attribute__((ext_vector_type(8)))  __bf16 v8bf;
typedef __attribute__((ext_vector_type(8)))  float  v8f;

// Problem constants (match reference)
#define BATCH 4
#define SEQ   2048
#define DIM   2048
#define HEADS 16
#define HD    128
#define RANK  16
#define NROW  (BATCH * SEQ)   // 8192

__device__ inline v16bf cat8(v8bf a, v8bf b) {
    return __builtin_shufflevector(a, b, 0,1,2,3,4,5,6,7,8,9,10,11,12,13,14,15);
}

__device__ inline v8f wmma_bf16(v16bf a, v16bf b, v8f c) {
    // D = A(16x32) * B(32x16) + C, fp32 accumulate
    return __builtin_amdgcn_wmma_f32_16x16x32_bf16(
        /*neg_a=*/false, a, /*neg_b=*/false, b,
        /*c_mod=*/(short)0, c, /*reuse_a=*/false, /*reuse_b=*/false);
}

// ---------------------------------------------------------------------------
// fp32 -> bf16 conversion (grid-stride)
// ---------------------------------------------------------------------------
__global__ void cvt_f32_bf16(const float* __restrict__ s, __bf16* __restrict__ d, int n) {
    int i = blockIdx.x * blockDim.x + threadIdx.x;
    int stride = gridDim.x * blockDim.x;
    for (; i < n; i += stride) d[i] = (__bf16)s[i];
}

// ---------------------------------------------------------------------------
// LoRA rank projection: T[n,r] = sum_k X[n,k] * A[r,k]   (X: [NROW,DIM] bf16,
// A: [RANK,DIM] bf16, T: [NROW,RANK] bf16). Tiny FLOP count; VALU is fine.
// ---------------------------------------------------------------------------
__global__ void lora_t_kernel(const __bf16* __restrict__ X,
                              const __bf16* __restrict__ A,
                              __bf16* __restrict__ T) {
    int g = blockIdx.x * blockDim.x + threadIdx.x;    // 0 .. NROW*RANK-1
    int n = g >> 4;
    int r = g & 15;
    const __bf16* xp = X + (size_t)n * DIM;
    const __bf16* ap = A + (size_t)r * DIM;
    float acc = 0.f;
    for (int k = 0; k < DIM; k += 8) {
        v8bf xv = *(const v8bf*)(xp + k);
        v8bf av = *(const v8bf*)(ap + k);
#pragma unroll
        for (int j = 0; j < 8; ++j) acc += (float)xv[j] * (float)av[j];
    }
    T[(size_t)n * RANK + r] = (__bf16)acc;
}

// ---------------------------------------------------------------------------
// WMMA GEMM with fused LoRA:  Y = X * W^T + T * BL^T
//   X : [NROW, DIM] bf16 row-major (A-matrix rows)
//   W : [DIM, DIM]  bf16 row-major (row j supplies B-matrix column j)
//   T : [NROW, RANK] bf16,  BL : [DIM, RANK] bf16 (rank-16 K padded to 32)
// Store modes: 0 = bf16 [B,H,S,HD] (Q,K)   1 = bf16 [B,H,HD,S] (V, transposed)
//              3 = fp32 row-major [NROW,DIM] (final output)
// Workgroup = 256 threads = 8 waves, WG tile 128x128, wave tile 32x64.
// ---------------------------------------------------------------------------
__global__ __launch_bounds__(256) void gemm_lora_kernel(
    const __bf16* __restrict__ X, const __bf16* __restrict__ W,
    const __bf16* __restrict__ T, const __bf16* __restrict__ BL,
    void* __restrict__ out, int mode) {

    const int tid  = threadIdx.x;
    const int lane = tid & 31;
    const int wave = tid >> 5;
    const int lh   = lane >> 4;      // lane half (0/1)
    const int ln   = lane & 15;      // lane index within half
    const int Mbase = blockIdx.x * 128 + (wave >> 1) * 32;
    const int Nbase = blockIdx.y * 128 + (wave & 1) * 64;

    const __bf16* arow[2];
#pragma unroll
    for (int ms = 0; ms < 2; ++ms)
        arow[ms] = X + (size_t)(Mbase + ms * 16 + ln) * DIM;
    const __bf16* brow[4];
#pragma unroll
    for (int ns = 0; ns < 4; ++ns)
        brow[ns] = W + (size_t)(Nbase + ns * 16 + ln) * DIM;

    v8f acc[2][4];
#pragma unroll
    for (int ms = 0; ms < 2; ++ms)
#pragma unroll
        for (int ns = 0; ns < 4; ++ns)
#pragma unroll
            for (int i = 0; i < 8; ++i) acc[ms][ns][i] = 0.f;

    for (int k0 = 0; k0 < DIM; k0 += 32) {
        v16bf af[2];
#pragma unroll
        for (int ms = 0; ms < 2; ++ms)
            af[ms] = cat8(*(const v8bf*)(arow[ms] + k0 + lh * 8),
                          *(const v8bf*)(arow[ms] + k0 + 16 + lh * 8));
        v16bf bf_[4];
#pragma unroll
        for (int ns = 0; ns < 4; ++ns)
            bf_[ns] = cat8(*(const v8bf*)(brow[ns] + k0 + lh * 8),
                           *(const v8bf*)(brow[ns] + k0 + 16 + lh * 8));
#pragma unroll
        for (int ms = 0; ms < 2; ++ms)
#pragma unroll
            for (int ns = 0; ns < 4; ++ns)
                acc[ms][ns] = wmma_bf16(af[ms], bf_[ns], acc[ms][ns]);
    }

    // ---- fused LoRA epilogue: one extra K=32 WMMA (upper 16 K zero) ----
    v8bf z;
#pragma unroll
    for (int j = 0; j < 8; ++j) z[j] = (__bf16)0.f;

    v16bf tf[2];
#pragma unroll
    for (int ms = 0; ms < 2; ++ms) {
        int row = Mbase + ms * 16 + ln;
        v8bf lo = *(const v8bf*)(T + (size_t)row * RANK + lh * 8);   // k = lh*8..lh*8+7 (<16)
        tf[ms] = cat8(lo, z);                                        // k>=16 -> 0
    }
    v16bf blf[4];
#pragma unroll
    for (int ns = 0; ns < 4; ++ns) {
        int col = Nbase + ns * 16 + ln;
        v8bf b0 = *(const v8bf*)(BL + (size_t)col * RANK + lh * 8);      // k = lh*8..lh*8+7
        v8bf b1 = z;                                                     // k>=16 -> 0
        blf[ns] = cat8(b0, b1);
    }
#pragma unroll
    for (int ms = 0; ms < 2; ++ms)
#pragma unroll
        for (int ns = 0; ns < 4; ++ns)
            acc[ms][ns] = wmma_bf16(tf[ms], blf[ns], acc[ms][ns]);

    // ---- store ----
    if (mode == 3) {
        float* o = (float*)out;
#pragma unroll
        for (int ms = 0; ms < 2; ++ms)
#pragma unroll
            for (int ns = 0; ns < 4; ++ns)
#pragma unroll
                for (int i = 0; i < 8; ++i) {
                    int r = Mbase + ms * 16 + i + 8 * lh;
                    int c = Nbase + ns * 16 + ln;
                    o[(size_t)r * DIM + c] = acc[ms][ns][i];
                }
    } else if (mode == 0) {            // bf16 [B,H,S,HD]
        __bf16* o = (__bf16*)out;
#pragma unroll
        for (int ms = 0; ms < 2; ++ms)
#pragma unroll
            for (int ns = 0; ns < 4; ++ns)
#pragma unroll
                for (int i = 0; i < 8; ++i) {
                    int r = Mbase + ms * 16 + i + 8 * lh;
                    int c = Nbase + ns * 16 + ln;
                    int b = r >> 11, s = r & 2047;
                    int h = c >> 7,  hd = c & 127;
                    o[(((size_t)(b * HEADS + h)) * SEQ + s) * HD + hd] = (__bf16)acc[ms][ns][i];
                }
    } else {                           // mode 1: bf16 [B,H,HD,S] (V transposed)
        __bf16* o = (__bf16*)out;
#pragma unroll
        for (int ms = 0; ms < 2; ++ms)
#pragma unroll
            for (int ns = 0; ns < 4; ++ns)
#pragma unroll
                for (int i = 0; i < 8; ++i) {
                    int r = Mbase + ms * 16 + i + 8 * lh;
                    int c = Nbase + ns * 16 + ln;
                    int b = r >> 11, s = r & 2047;
                    int h = c >> 7,  hd = c & 127;
                    o[(((size_t)(b * HEADS + h)) * HD + hd) * SEQ + s] = (__bf16)acc[ms][ns][i];
                }
    }
}

// ---------------------------------------------------------------------------
// Flash attention (causal), one wave per 16-row q tile, kv in blocks of 32.
//   Q,K : bf16 [B,H,S,HD]     V : bf16 [B,H,HD,S]    O : bf16 [NROW,DIM]
// Score tiles via WMMA (contraction over HD, 4 chunks of 32).
// P is written through LDS to convert C-layout -> A-layout, then 8 PV WMMAs.
// ---------------------------------------------------------------------------
__global__ __launch_bounds__(128) void flash_attn_kernel(
    const __bf16* __restrict__ Q, const __bf16* __restrict__ K,
    const __bf16* __restrict__ V, __bf16* __restrict__ O) {

    __shared__ __align__(16) __bf16 Pls[4][16 * 32];

    const int lane = threadIdx.x & 31;
    const int w    = threadIdx.x >> 5;
    const int task = blockIdx.x * 4 + w;        // B*H*(S/16) = 8192 tasks
    const int qt   = task & 127;
    const int bh   = task >> 7;
    const int q0   = qt * 16;
    const int lh   = lane >> 4;
    const int ln   = lane & 15;

    const __bf16* Qp = Q + (size_t)bh * SEQ * HD;
    const __bf16* Kp = K + (size_t)bh * SEQ * HD;
    const __bf16* Vp = V + (size_t)bh * HD * SEQ;
    __bf16* myP = &Pls[w][0];

    // Q fragments: 4 chunks over HD (A-matrix, 16x32 each)
    v16bf qf[4];
    {
        const __bf16* qrow = Qp + (size_t)(q0 + ln) * HD;
#pragma unroll
        for (int c = 0; c < 4; ++c)
            qf[c] = cat8(*(const v8bf*)(qrow + c * 32 + lh * 8),
                         *(const v8bf*)(qrow + c * 32 + 16 + lh * 8));
    }

    float mstate[8], lstate[8];
    v8f acc[8];
#pragma unroll
    for (int i = 0; i < 8; ++i) { mstate[i] = -1e30f; lstate[i] = 0.f; }
#pragma unroll
    for (int c = 0; c < 8; ++c)
#pragma unroll
        for (int i = 0; i < 8; ++i) acc[c][i] = 0.f;

    const float scale = 0.08838834764831845f;   // 1/sqrt(128)

    for (int j0 = 0; j0 <= q0 + 15; j0 += 32) {
        // ---- scores: S = Q * K^T, two 16x16 tiles (kv j0..+15, j0+16..+31) ----
        v8f s0, s1;
#pragma unroll
        for (int i = 0; i < 8; ++i) { s0[i] = 0.f; s1[i] = 0.f; }
#pragma unroll
        for (int c = 0; c < 4; ++c) {
            const __bf16* k0p = Kp + (size_t)(j0 + ln) * HD + c * 32 + lh * 8;
            const __bf16* k1p = Kp + (size_t)(j0 + 16 + ln) * HD + c * 32 + lh * 8;
            v16bf kb0 = cat8(*(const v8bf*)k0p, *(const v8bf*)(k0p + 16));
            v16bf kb1 = cat8(*(const v8bf*)k1p, *(const v8bf*)(k1p + 16));
            s0 = wmma_bf16(qf[c], kb0, s0);
            s1 = wmma_bf16(qf[c], kb1, s1);
        }

        // ---- causal mask + online softmax (row stats across 16-lane groups) ----
        float p0[8], p1[8], fac[8];
#pragma unroll
        for (int i = 0; i < 8; ++i) {
            int m  = i + 8 * lh;
            int qg = q0 + m;
            float v0 = (j0 + ln      <= qg) ? s0[i] * scale : -1e30f;
            float v1 = (j0 + 16 + ln <= qg) ? s1[i] * scale : -1e30f;
            float rm = fmaxf(v0, v1);
            rm = fmaxf(rm, __shfl_xor(rm, 1));
            rm = fmaxf(rm, __shfl_xor(rm, 2));
            rm = fmaxf(rm, __shfl_xor(rm, 4));
            rm = fmaxf(rm, __shfl_xor(rm, 8));
            float mnew = fmaxf(mstate[i], rm);
            fac[i] = __expf(mstate[i] - mnew);
            mstate[i] = mnew;
            p0[i] = __expf(v0 - mnew);
            p1[i] = __expf(v1 - mnew);
            float rs = p0[i] + p1[i];
            rs += __shfl_xor(rs, 1);
            rs += __shfl_xor(rs, 2);
            rs += __shfl_xor(rs, 4);
            rs += __shfl_xor(rs, 8);
            lstate[i] = lstate[i] * fac[i] + rs;
        }
#pragma unroll
        for (int c = 0; c < 8; ++c)
#pragma unroll
            for (int i = 0; i < 8; ++i) acc[c][i] *= fac[i];

        // ---- P: C-layout -> A-layout via LDS (per-wave private 16x32 tile) ----
#pragma unroll
        for (int i = 0; i < 8; ++i) {
            int m = i + 8 * lh;
            myP[m * 32 + ln]      = (__bf16)p0[i];
            myP[m * 32 + 16 + ln] = (__bf16)p1[i];
        }
        v16bf pa = cat8(*(const v8bf*)(myP + ln * 32 + lh * 8),
                        *(const v8bf*)(myP + ln * 32 + 16 + lh * 8));

        // ---- O += P * V  (V transposed: contiguous along kv) ----
#pragma unroll
        for (int c = 0; c < 8; ++c) {
            const __bf16* vp = Vp + (size_t)(c * 16 + ln) * SEQ + j0 + lh * 8;
            v16bf vb = cat8(*(const v8bf*)vp, *(const v8bf*)(vp + 16));
            acc[c] = wmma_bf16(pa, vb, acc[c]);
        }
    }

    // ---- epilogue: O / l, store bf16 row-major [NROW, DIM] ----
    const int b = bh >> 4, h = bh & 15;
#pragma unroll
    for (int i = 0; i < 8; ++i) {
        float inv = 1.f / lstate[i];
        int m = i + 8 * lh;
        int s = q0 + m;
#pragma unroll
        for (int c = 0; c < 8; ++c) {
            int col = h * HD + c * 16 + ln;
            O[((size_t)(b * SEQ + s)) * DIM + col] = (__bf16)(acc[c][i] * inv);
        }
    }
}

// ---------------------------------------------------------------------------
// Host-side orchestration
// ---------------------------------------------------------------------------
extern "C" void kernel_launch(void* const* d_in, const int* in_sizes, int n_in,
                              void* d_out, int out_size, void* d_ws, size_t ws_size,
                              hipStream_t stream) {
    (void)in_sizes; (void)n_in; (void)out_size; (void)ws_size;
    // d_in order: x, mask, Wq, Aq, Bq, Wk, Ak, Bk, Wv, Av, Bv, Wo, Ao, Bo
    const float* x = (const float*)d_in[0];
    const float* Wf[4] = { (const float*)d_in[2], (const float*)d_in[5],
                           (const float*)d_in[8], (const float*)d_in[11] };
    const float* Af[4] = { (const float*)d_in[3], (const float*)d_in[6],
                           (const float*)d_in[9], (const float*)d_in[12] };
    const float* Bf[4] = { (const float*)d_in[4], (const float*)d_in[7],
                           (const float*)d_in[10], (const float*)d_in[13] };

    char* ws = (char*)d_ws;
    size_t off = 0;
    auto take = [&](size_t bytes) { char* p = ws + off; off += bytes; return p; };

    __bf16* xb = (__bf16*)take((size_t)NROW * DIM * 2);        // 32 MB
    __bf16* Wb[4], *Ab[4], *Blb[4], *Tb[4];
    for (int i = 0; i < 4; ++i) Wb[i]  = (__bf16*)take((size_t)DIM * DIM * 2);   // 4 x 8 MB
    for (int i = 0; i < 4; ++i) Ab[i]  = (__bf16*)take((size_t)RANK * DIM * 2);
    for (int i = 0; i < 4; ++i) Blb[i] = (__bf16*)take((size_t)DIM * RANK * 2);
    for (int i = 0; i < 4; ++i) Tb[i]  = (__bf16*)take((size_t)NROW * RANK * 2);
    __bf16* Qb = (__bf16*)take((size_t)NROW * DIM * 2);        // [B,H,S,HD]
    __bf16* Kb = (__bf16*)take((size_t)NROW * DIM * 2);        // [B,H,S,HD]
    __bf16* Vb = (__bf16*)take((size_t)NROW * DIM * 2);        // [B,H,HD,S]
    __bf16* Ob = (__bf16*)take((size_t)NROW * DIM * 2);        // [NROW,DIM]

    auto cvt = [&](const float* s, __bf16* d, int n) {
        int blocks = (n + 255) / 256; if (blocks > 4096) blocks = 4096;
        cvt_f32_bf16<<<blocks, 256, 0, stream>>>(s, d, n);
    };

    cvt(x, xb, NROW * DIM);
    for (int i = 0; i < 4; ++i) cvt(Wf[i], Wb[i], DIM * DIM);
    for (int i = 0; i < 4; ++i) cvt(Af[i], Ab[i], RANK * DIM);
    for (int i = 0; i < 4; ++i) cvt(Bf[i], Blb[i], DIM * RANK);

    dim3 ggrid(NROW / 128, DIM / 128, 1);

    // Q projection
    lora_t_kernel<<<(NROW * RANK) / 256, 256, 0, stream>>>(xb, Ab[0], Tb[0]);
    gemm_lora_kernel<<<ggrid, 256, 0, stream>>>(xb, Wb[0], Tb[0], Blb[0], Qb, 0);
    // K projection
    lora_t_kernel<<<(NROW * RANK) / 256, 256, 0, stream>>>(xb, Ab[1], Tb[1]);
    gemm_lora_kernel<<<ggrid, 256, 0, stream>>>(xb, Wb[1], Tb[1], Blb[1], Kb, 0);
    // V projection (transposed store)
    lora_t_kernel<<<(NROW * RANK) / 256, 256, 0, stream>>>(xb, Ab[2], Tb[2]);
    gemm_lora_kernel<<<ggrid, 256, 0, stream>>>(xb, Wb[2], Tb[2], Blb[2], Vb, 1);

    // attention: 8192 wave-tasks, 4 waves per block
    flash_attn_kernel<<<2048, 128, 0, stream>>>(Qb, Kb, Vb, Ob);

    // output projection -> fp32 d_out
    lora_t_kernel<<<(NROW * RANK) / 256, 256, 0, stream>>>(Ob, Ab[3], Tb[3]);
    gemm_lora_kernel<<<ggrid, 256, 0, stream>>>(Ob, Wb[3], Tb[3], Blb[3], d_out, 3);
}
